// BayesianKAN_ECG_88098369175724
// MI455X (gfx1250) — matrix operand order repaired and checked by reading, not compiled, not run on hardware
//
#include <hip/hip_runtime.h>

typedef __attribute__((ext_vector_type(16))) _Float16 v16h;
typedef __attribute__((ext_vector_type(8)))  float    v8f;

#define SEQ_LEN 1000
#define IN_DIM  100
#define POOLW   10
#define HIDDEN  64
#define OUT_DIM 5
#define NB      16
#define K1      (IN_DIM * NB)   // 1600
#define K2      (HIDDEN * NB)   // 1024
#define ROWS_WG 128             // 8 waves * 16 rows
// -0.5/0.36 * log2(e): RBF exponent rebased to exp2 for native v_exp_f32
#define RBF_C2  (-2.0037431123457564f)

#if __has_builtin(__builtin_amdgcn_exp2f)
#define EXP2F(v) __builtin_amdgcn_exp2f(v)
#else
#define EXP2F(v) exp2f(v)
#endif
#if __has_builtin(__builtin_amdgcn_rcpf)
#define RCPF(v) __builtin_amdgcn_rcpf(v)
#else
#define RCPF(v) (1.0f / (v))
#endif

// Branch-free tanh on native exp2/rcp: exp2 overflow -> inf -> rcp -> 0 -> result 1.
__device__ __forceinline__ float fast_tanh(float x) {
    float ax = __builtin_fabsf(x);
    float e  = EXP2F(2.885390081777927f * ax);     // exp(2|x|)
    float r  = 1.f - 2.f * RCPF(e + 1.f);
    return __builtin_copysignf(r, x);
}

__device__ __forceinline__ v16h load_b_f16(const _Float16* __restrict__ p) {
    typedef __attribute__((ext_vector_type(8))) _Float16 v8h_t;
    v8h_t lo = *(const v8h_t*)p, hi = *(const v8h_t*)(p + 16);   // 16B aligned -> 2x b128 (p[0..7], p[16..23])
    return __builtin_shufflevector(lo, hi, 0,1,2,3,4,5,6,7,8,9,10,11,12,13,14,15);
}

__device__ __forceinline__ v16h load_b_f32cvt(const float* __restrict__ p) {
    const float4* wp = (const float4*)p;
    float4 q0 = wp[0], q1 = wp[1], q2 = wp[4], q3 = wp[5];
    v16h b;
    b[0]=(_Float16)q0.x;  b[1]=(_Float16)q0.y;  b[2]=(_Float16)q0.z;  b[3]=(_Float16)q0.w;
    b[4]=(_Float16)q1.x;  b[5]=(_Float16)q1.y;  b[6]=(_Float16)q1.z;  b[7]=(_Float16)q1.w;
    b[8]=(_Float16)q2.x;  b[9]=(_Float16)q2.y;  b[10]=(_Float16)q2.z; b[11]=(_Float16)q2.w;
    b[12]=(_Float16)q3.x; b[13]=(_Float16)q3.y; b[14]=(_Float16)q3.z; b[15]=(_Float16)q3.w;
    return b;
}

// Issue one x-row (1000 f32 = 250 x 16B) as 8 per-lane async global->LDS b128 loads.
// Generic pointers to __shared__ carry the LDS offset in their low 32 bits (FLAT aperture).
__device__ __forceinline__ void async_row(const float* __restrict__ gp, uint32_t lds_base, int lane) {
#pragma unroll
    for (int u = 0; u < 8; ++u) {
        int t = lane + 32 * u; t = (t < 250) ? t : 249;     // clamped dup: same data, benign
        unsigned long long ga = (unsigned long long)(uintptr_t)gp + (unsigned long long)(t * 16);
        uint32_t la = lds_base + (uint32_t)(t * 16);
        asm volatile("global_load_async_to_lds_b128 %0, %1, off"
                     :: "v"(la), "v"(ga) : "memory");
    }
}

// f32 -> f16 weight conversion into workspace (once per launch, layout unchanged)
__global__ __launch_bounds__(256)
void cvt_weights_kernel(const float* __restrict__ c1, const float* __restrict__ c2,
                        _Float16* __restrict__ w) {
    int i = blockIdx.x * 256 + threadIdx.x;
    if (i < K1 * HIDDEN)                     w[i] = (_Float16)c1[i];
    else if (i < K1 * HIDDEN + K2 * OUT_DIM) w[i] = (_Float16)c2[i - K1 * HIDDEN];
}

template<bool WF16>
__global__ __launch_bounds__(256)
void kan_fused_kernel(const float* __restrict__ x,
                      const void* __restrict__ w1v,
                      const void* __restrict__ w2v,
                      const float* __restrict__ centers,
                      float* __restrict__ out)
{
    __shared__ float stage[8][2][SEQ_LEN];       // 64 KB  per-wave double-buffered row staging
    __shared__ float xn[ROWS_WG][IN_DIM];        // 51.2 KB normalized pooled x
    __shared__ float hbuf[ROWS_WG][HIDDEN + 4];  // 34.8 KB tanh(h), padded vs bank conflicts

    const int tid   = threadIdx.x;
    const int wave  = tid >> 5;
    const int lane  = tid & 31;
    const int m     = lane & 15;     // row within M-tile (A) or col within N-tile (B/C)
    const int g     = lane >> 4;     // half-wave group
    const int rl0   = wave * 16;
    const long rg0  = (long)blockIdx.x * ROWS_WG + rl0;

    // Each half-wave only uses centers[8g..8g+7]; expand C2*(x-c)^2 = fma(x,beta,gamma) + C2*x^2.
    float beta[8], gamma[8];
#pragma unroll
    for (int j = 0; j < 8; ++j) {
        float c = centers[g * 8 + j];
        beta[j]  = -2.f * RBF_C2 * c;
        gamma[j] = RBF_C2 * c * c;
    }

    const uint32_t stage_lds0 = (uint32_t)(uintptr_t)&stage[wave][0][0];
    const uint32_t stage_lds1 = (uint32_t)(uintptr_t)&stage[wave][1][0];

    // ---------------- Phase A: avg-pool 1000->100 and normalize (ddof=1) ----------------
    // Async-pipelined: row rr+1 streams into the other LDS buffer while row rr is pooled.
    async_row(x + rg0 * (long)SEQ_LEN, stage_lds0, lane);
    for (int rr = 0; rr < 16; ++rr) {
        const int buf = rr & 1;
        if (rr < 15)
            async_row(x + (rg0 + rr + 1) * (long)SEQ_LEN, (rr & 1) ? stage_lds0 : stage_lds1, lane);
        if (rr < 15) asm volatile("s_wait_asynccnt 0x8" ::: "memory");  // row rr landed (in-order)
        else         asm volatile("s_wait_asynccnt 0x0" ::: "memory");

        auto poolsum = [&](int i) {
            const float2* p2 = (const float2*)&stage[wave][buf][i * POOLW];
            float a = 0.f;
#pragma unroll
            for (int qq = 0; qq < 5; ++qq) { float2 v = p2[qq]; a += v.x + v.y; }
            return a * 0.1f;
        };
        float pv0 = poolsum(lane);
        float pv1 = poolsum(lane + 32);
        float pv2 = poolsum(lane + 64);
        float pv3 = (lane < 4) ? poolsum(lane + 96) : 0.f;

        float s  = pv0 + pv1 + pv2 + pv3;
        float s2 = pv0*pv0 + pv1*pv1 + pv2*pv2 + pv3*pv3;
#pragma unroll
        for (int off = 16; off >= 1; off >>= 1) {
            s  += __shfl_xor(s,  off, 32);
            s2 += __shfl_xor(s2, off, 32);
        }
        float mean = s * (1.f / IN_DIM);
        float var  = (s2 - (float)IN_DIM * mean * mean) * (1.f / (IN_DIM - 1));
        float inv  = 1.f / (sqrtf(fmaxf(var, 0.f)) + 1e-6f);

        float* xr = xn[rl0 + rr];
        xr[lane]      = (pv0 - mean) * inv;
        xr[lane + 32] = (pv1 - mean) * inv;
        xr[lane + 64] = (pv2 - mean) * inv;
        if (lane < 4) xr[lane + 96] = (pv3 - mean) * inv;
    }
    asm volatile("s_wait_dscnt 0x0" ::: "memory");   // xn visible to this wave

    const int myrow = rl0 + m;

    // ---------------- Phase B: GEMM1  h = RBF(xn)[16x1600] @ W1^T[1600x64] ----------------
    v8f h0 = {}, h1 = {}, h2 = {}, h3 = {};
#pragma unroll 1
    for (int kb = 0; kb < K1 / 32; ++kb) {
        const float2 xv = *(const float2*)&xn[myrow][2 * kb];   // i = 2kb, 2kb+1 (same for both halves)
        const float P0 = RBF_C2 * xv.x * xv.x;
        const float P1 = RBF_C2 * xv.y * xv.y;
        v16h a;
#pragma unroll
        for (int p = 0; p < 8; ++p)
            a[p]     = (_Float16)EXP2F(__builtin_fmaf(xv.x, beta[p], gamma[p]) + P0);
#pragma unroll
        for (int p = 0; p < 8; ++p)
            a[8 + p] = (_Float16)EXP2F(__builtin_fmaf(xv.y, beta[p], gamma[p]) + P1);

        const int k0 = kb * 32 + g * 8;    // B: half g K=8g..8g+7 and 16+8g..16+8g+7
#pragma unroll
        for (int nt = 0; nt < 4; ++nt) {
            v16h b;
            if (WF16) b = load_b_f16((const _Float16*)w1v + (nt * 16 + m) * K1 + k0);
            else      b = load_b_f32cvt((const float*)w1v + (nt * 16 + m) * K1 + k0);
            if      (nt == 0) h0 = __builtin_amdgcn_wmma_f32_16x16x32_f16(false, a, false, b, (short)0, h0, false, false);
            else if (nt == 1) h1 = __builtin_amdgcn_wmma_f32_16x16x32_f16(false, a, false, b, (short)0, h1, false, false);
            else if (nt == 2) h2 = __builtin_amdgcn_wmma_f32_16x16x32_f16(false, a, false, b, (short)0, h2, false, false);
            else              h3 = __builtin_amdgcn_wmma_f32_16x16x32_f16(false, a, false, b, (short)0, h3, false, false);
        }
    }

    // ---------------- Phase C: tanh -> hbuf ----------------
#pragma unroll
    for (int r = 0; r < 8; ++r) {
        const int rl = rl0 + g * 8 + r;              // C layout: lane group g holds rows g*8..g*8+7
        hbuf[rl][ 0 + m] = fast_tanh(h0[r]);
        hbuf[rl][16 + m] = fast_tanh(h1[r]);
        hbuf[rl][32 + m] = fast_tanh(h2[r]);
        hbuf[rl][48 + m] = fast_tanh(h3[r]);
    }
    asm volatile("s_wait_dscnt 0x0" ::: "memory");

    // ---------------- Phase D: per-row stats of h (ddof=1); lane's row == myrow ----------------
    float s = 0.f, s2 = 0.f;
    const float4* hr = (const float4*)&hbuf[myrow][g * 32];
#pragma unroll
    for (int c = 0; c < 8; ++c) {
        float4 v = hr[c];
        s  += v.x + v.y + v.z + v.w;
        s2 += v.x*v.x + v.y*v.y + v.z*v.z + v.w*v.w;
    }
    s  += __shfl_xor(s,  16, 32);
    s2 += __shfl_xor(s2, 16, 32);
    const float hmean = s * (1.f / HIDDEN);
    const float hvar  = (s2 - (float)HIDDEN * hmean * hmean) * (1.f / (HIDDEN - 1));
    const float hinv  = 1.f / (sqrtf(fmaxf(hvar, 0.f)) + 1e-6f);

    // ---------------- Phase E: GEMM2  out = RBF(norm(h))[16x1024] @ W2^T[1024x5] ----------------
    v8f o = {};
    const bool bvalid = (m < OUT_DIM);
#pragma unroll 1
    for (int kb = 0; kb < K2 / 32; ++kb) {
        const float2 hv2 = *(const float2*)&hbuf[myrow][2 * kb];
        const float hv0 = (hv2.x - hmean) * hinv;
        const float hv1 = (hv2.y - hmean) * hinv;
        const float P0 = RBF_C2 * hv0 * hv0;
        const float P1 = RBF_C2 * hv1 * hv1;
        v16h a;
#pragma unroll
        for (int p = 0; p < 8; ++p)
            a[p]     = (_Float16)EXP2F(__builtin_fmaf(hv0, beta[p], gamma[p]) + P0);
#pragma unroll
        for (int p = 0; p < 8; ++p)
            a[8 + p] = (_Float16)EXP2F(__builtin_fmaf(hv1, beta[p], gamma[p]) + P1);

        const int k0 = kb * 32 + g * 8;
        v16h b = {};
        if (bvalid) {
            if (WF16) b = load_b_f16((const _Float16*)w2v + m * K2 + k0);
            else      b = load_b_f32cvt((const float*)w2v + m * K2 + k0);
        }
        o = __builtin_amdgcn_wmma_f32_16x16x32_f16(false, a, false, b, (short)0, o, false, false);
    }

    // ---------------- Store: C tile rows g*8+r, col m (only m < 5 valid) ----------------
    if (bvalid) {
        float* op = out + (rg0 + g * 8) * OUT_DIM + m;
#pragma unroll
        for (int r = 0; r < 8; ++r) op[r * OUT_DIM] = o[r];
    }
}

extern "C" void kernel_launch(void* const* d_in, const int* in_sizes, int n_in,
                              void* d_out, int out_size, void* d_ws, size_t ws_size,
                              hipStream_t stream) {
    const float* x       = (const float*)d_in[0];
    const float* c1      = (const float*)d_in[1];
    const float* c2      = (const float*)d_in[2];
    const float* centers = (const float*)d_in[3];
    float* out = (float*)d_out;

    const int batch  = in_sizes[0] / SEQ_LEN;       // 65536
    const int blocks = batch / ROWS_WG;             // 512

    const size_t wneed = (size_t)(K1 * HIDDEN + K2 * OUT_DIM) * sizeof(_Float16); // 215040 B
    if (ws_size >= wneed) {
        _Float16* w = (_Float16*)d_ws;
        const int total = K1 * HIDDEN + K2 * OUT_DIM;
        hipLaunchKernelGGL(cvt_weights_kernel, dim3((total + 255) / 256), dim3(256), 0, stream,
                           c1, c2, w);
        hipLaunchKernelGGL((kan_fused_kernel<true>), dim3(blocks), dim3(256), 0, stream,
                           x, (const void*)w, (const void*)(w + K1 * HIDDEN), centers, out);
    } else {
        hipLaunchKernelGGL((kan_fused_kernel<false>), dim3(blocks), dim3(256), 0, stream,
                           x, (const void*)c1, (const void*)c2, centers, out);
    }
}
